// KernelLookup_22393959481895
// MI455X (gfx1250) — matrix operand order repaired and checked by reading, not compiled, not run on hardware
//
#include <hip/hip_runtime.h>

typedef __attribute__((ext_vector_type(16))) _Float16 v16h;
typedef __attribute__((ext_vector_type(8)))  float    v8f;
typedef __attribute__((ext_vector_type(4)))  float    v4f;

#define C_IN   16
#define KSZ    3
#define KK     9          // KSZ*KSZ
#define KLEN   144        // C_IN*KK; padded to 160 = 5*32
#define HALFT  72         // KLEN/2 = 8 channels * 9 taps
#define DH     128
#define DW     128
#define HO     126
#define WO     126
#define NBATCH 8

// One wave -> 16 consecutive output pixels of one row, via the diagonal of a
// 16x16 WMMA tile. Reduction slots are permuted so that:
//   hf=0 lanes carry t = 16*it + j         (channels 0..7)
//   hf=1 lanes carry t = 16*it + j + 72    (channels 8..15, constant +8*H*W)
// => all A gather offsets are compile-time immediates (no div/mod VALU),
// => each lane's B fragment is two contiguous 32B runs of the selected
//    144-float weight vector (global_load_b128 with immediate offsets).
__global__ __launch_bounds__(128)
void KernelLookup_wmma(const float* __restrict__ data,
                       const int*   __restrict__ kidx,
                       const float* __restrict__ weights,
                       float*       __restrict__ out)
{
    const int lane = threadIdx.x & 31;
    const int wave = threadIdx.x >> 5;
    const int h    = blockIdx.y;              // 0..125
    const int b    = blockIdx.z;              // 0..7
    const int w0   = blockIdx.x * 64 + wave * 16;

    const int pix  = lane & 15;               // pixel within 16-wide tile
    const int hf   = lane >> 4;               // half-wave selector

    // Clamp tail pixels (w=126,127 of the last tile): lanes still feed the
    // WMMA (EXEC must be all-ones) but never store.
    int wp = w0 + pix;
    int wc = wp < (WO - 1) ? wp : (WO - 1);

    // Per-pixel kernel id; its weight vector = 144 contiguous floats.
    const int sel = kidx[(b * HO + h) * WO + wc];
    const float* __restrict__ wvec = weights + (size_t)sel * KLEN;

    // Per-lane base pointers; hf folds into a constant channel offset.
    const float* __restrict__ pA =
        data + (size_t)b * C_IN * DH * DW + h * DW + wc + hf * (8 * DH * DW);
    const float* __restrict__ pW = wvec + hf * HALFT;

    v8f acc = {};

    #pragma unroll
    for (int it = 0; it < 5; ++it) {
        // ---- A fragment: element j holds t = 16*it + j (+72 if hf) ----
        v16h a;
        #pragma unroll
        for (int j = 0; j < 16; ++j) {
            const int tj = 16 * it + j;            // compile-time after unroll
            if (tj < HALFT) {
                const int c  = tj / KK;            // folds to immediates
                const int r  = tj - KK * c;
                const int dy = r / KSZ;
                const int dx = r - KSZ * dy;
                a[j] = (_Float16)pA[(c * DH + dy) * DW + dx];
            } else {
                a[j] = (_Float16)0.0f;             // K padding (it==4, j>=8)
            }
        }

        // ---- B fragment: one contiguous 16-float run of wvec ----
        // elements 0..15 = wvec[hf*72 + 16*it + 0..15] (8..15 zero when it==4)
        v16h bm = {};
        {
            const v4f* __restrict__ q0 = (const v4f*)(pW + 16 * it);
            v4f f0 = q0[0], f1 = q0[1];
            #pragma unroll
            for (int e = 0; e < 4; ++e) {
                bm[e +  0] = (_Float16)f0[e];
                bm[e +  4] = (_Float16)f1[e];
            }
            if (it < 4) {                          // it==4, elements 8..15 are padding
                v4f f2 = q0[2], f3 = q0[3];
                #pragma unroll
                for (int e = 0; e < 4; ++e) {
                    bm[e +  8] = (_Float16)f2[e];
                    bm[e + 12] = (_Float16)f3[e];
                }
            }
        }

        // D = A x B + acc   (v_wmma_f32_16x16x32_f16)
        acc = __builtin_amdgcn_wmma_f32_16x16x32_f16(
            /*neg_a=*/false, a, /*neg_b=*/false, bm,
            /*c_mod=*/(short)0, acc, /*reuse_a=*/false, /*reuse_b=*/false);
    }

    // ---- Harvest the diagonal of the 16x16 f32 accumulator ----
    // lanes 0-7   : D[m][m], m=lane,    in acc[lane]
    // lanes 24-31 : D[m][m], m=lane-16, in acc[lane-24]
    int vsel = -1;
    if (lane < 8)        vsel = lane;
    else if (lane >= 24) vsel = lane - 24;

    float val = 0.0f;
    #pragma unroll
    for (int j = 0; j < 8; ++j)
        if (vsel == j) val = acc[j];

    const int opix = (lane < 8) ? (w0 + lane) : (w0 + lane - 16);
    if (vsel >= 0 && opix < WO)
        out[(b * HO + h) * WO + opix] = val;
}

extern "C" void kernel_launch(void* const* d_in, const int* in_sizes, int n_in,
                              void* d_out, int out_size, void* d_ws, size_t ws_size,
                              hipStream_t stream)
{
    const float* data    = (const float*)d_in[0];  // [8,16,128,128] f32
    const int*   kidx    = (const int*)  d_in[1];  // [8,126,126] int
    const float* weights = (const float*)d_in[2];  // [128,16,3,3] f32
    float*       out     = (float*)d_out;          // [8,126,126] f32

    dim3 grid((WO + 63) / 64, HO, NBATCH);         // (2, 126, 8)
    dim3 block(128);                               // 4 waves -> 64 pixels/row
    hipLaunchKernelGGL(KernelLookup_wmma, grid, block, 0, stream,
                       data, kidx, weights, out);
}
